// MultiHeadAttention_63359357550984
// MI455X (gfx1250) — matrix operand order repaired and checked by reading, not compiled, not run on hardware
//
#include <hip/hip_runtime.h>
#include <stdint.h>

#define D_MODEL 1024
#define NHEADS  16
#define DK      64
#define BATCH   2
#define SEQ     2048
#define MROWS   (BATCH * SEQ)   // 4096 rows for all GEMMs

typedef __attribute__((ext_vector_type(16))) __bf16 v16bf;
typedef __attribute__((ext_vector_type(8)))  float  v8f;

union FragAB {
    v16bf v;
    uint4 q[2];
    unsigned short u[16];
};

__device__ __forceinline__ unsigned short f2bf(float f) {
    union { float f; unsigned int u; } c; c.f = f;
    unsigned int u = c.u;
    unsigned int r = u + 0x7FFFu + ((u >> 16) & 1u);   // round-to-nearest-even
    r = ((u & 0x7F800000u) == 0x7F800000u) ? u : r;    // inf/nan: truncate
    return (unsigned short)(r >> 16);
}

// ---------------------------------------------------------------- prep kernels
__global__ void cast_f32_bf16(const float* __restrict__ in,
                              unsigned short* __restrict__ out, int n) {
    int i = blockIdx.x * blockDim.x + threadIdx.x;
    if (i < n) out[i] = f2bf(in[i]);
}

// w: [K=1024][N=1024] fp32 -> wt: [N][K] bf16 (so WMMA B-frag loads are contiguous)
__global__ void transpose_cast(const float* __restrict__ w,
                               unsigned short* __restrict__ wt) {
    int i = blockIdx.x * blockDim.x + threadIdx.x;   // 1M elements
    int n = i & (D_MODEL - 1);
    int k = i >> 10;
    wt[(size_t)n * D_MODEL + k] = f2bf(w[(size_t)k * D_MODEL + n]);
}

// ---------------------------------------------------------------- WMMA GEMM
// C[4096][1024] = A[4096][1024](bf16) @ W(Wt is [col][k] bf16) + bias
// One wave computes a 32x64 block (2x4 WMMA tiles): 12 b128 loads per 8 WMMAs.
// mode 0: fp32 out [row][col]          (final projection)
// mode 1: bf16 out [b][h][s][d]        (Q, K)
// mode 2: bf16 out [b][h][d][s]        (V, d-major for contiguous PV B-frags)
__global__ void gemm_wmma(const unsigned short* __restrict__ A,
                          const unsigned short* __restrict__ Wt,
                          const float* __restrict__ bias,
                          float* __restrict__ outF,
                          unsigned short* __restrict__ outH,
                          int mode) {
    const int lane   = threadIdx.x & 31;
    const int waveId = blockIdx.x * (blockDim.x >> 5) + (threadIdx.x >> 5);
    const int mTile  = waveId >> 4;          // 1024/64 = 16 n-blocks
    const int nTile  = waveId & 15;
    const int m0 = mTile * 32, n0 = nTile * 64;
    const int hh = lane >> 4, nn = lane & 15;

    const unsigned short* aRow0 = A + (size_t)(m0 + nn) * D_MODEL;
    const unsigned short* aRow1 = A + (size_t)(m0 + 16 + nn) * D_MODEL;
    const unsigned short* bRow0 = Wt + (size_t)(n0 + nn) * D_MODEL;
    const unsigned short* bRow1 = Wt + (size_t)(n0 + 16 + nn) * D_MODEL;
    const unsigned short* bRow2 = Wt + (size_t)(n0 + 32 + nn) * D_MODEL;
    const unsigned short* bRow3 = Wt + (size_t)(n0 + 48 + nn) * D_MODEL;

    v8f acc[2][4] = {};
    #pragma unroll 4
    for (int kb = 0; kb < D_MODEL; kb += 32) {
        FragAB a0, a1, b0, b1, b2, b3;
        // A-frags: lane row m = nn (+16); k = hh*16 + e (contiguous 16 halves)
        a0.q[0] = *(const uint4*)(aRow0 + kb + hh * 16);
        a0.q[1] = *(const uint4*)(aRow0 + kb + hh * 16 + 8);
        a1.q[0] = *(const uint4*)(aRow1 + kb + hh * 16);
        a1.q[1] = *(const uint4*)(aRow1 + kb + hh * 16 + 8);
        // B-frags: lane col n = nn; k = hh*16 + e (contiguous 16 halves)
        b0.q[0] = *(const uint4*)(bRow0 + kb + hh * 16);
        b0.q[1] = *(const uint4*)(bRow0 + kb + hh * 16 + 8);
        b1.q[0] = *(const uint4*)(bRow1 + kb + hh * 16);
        b1.q[1] = *(const uint4*)(bRow1 + kb + hh * 16 + 8);
        b2.q[0] = *(const uint4*)(bRow2 + kb + hh * 16);
        b2.q[1] = *(const uint4*)(bRow2 + kb + hh * 16 + 8);
        b3.q[0] = *(const uint4*)(bRow3 + kb + hh * 16);
        b3.q[1] = *(const uint4*)(bRow3 + kb + hh * 16 + 8);

        acc[0][0] = __builtin_amdgcn_wmma_f32_16x16x32_bf16(false, a0.v, false, b0.v, (short)0, acc[0][0], false, false);
        acc[1][0] = __builtin_amdgcn_wmma_f32_16x16x32_bf16(false, a1.v, false, b0.v, (short)0, acc[1][0], false, false);
        acc[0][1] = __builtin_amdgcn_wmma_f32_16x16x32_bf16(false, a0.v, false, b1.v, (short)0, acc[0][1], false, false);
        acc[1][1] = __builtin_amdgcn_wmma_f32_16x16x32_bf16(false, a1.v, false, b1.v, (short)0, acc[1][1], false, false);
        acc[0][2] = __builtin_amdgcn_wmma_f32_16x16x32_bf16(false, a0.v, false, b2.v, (short)0, acc[0][2], false, false);
        acc[1][2] = __builtin_amdgcn_wmma_f32_16x16x32_bf16(false, a1.v, false, b2.v, (short)0, acc[1][2], false, false);
        acc[0][3] = __builtin_amdgcn_wmma_f32_16x16x32_bf16(false, a0.v, false, b3.v, (short)0, acc[0][3], false, false);
        acc[1][3] = __builtin_amdgcn_wmma_f32_16x16x32_bf16(false, a1.v, false, b3.v, (short)0, acc[1][3], false, false);
    }

    // epilogue: bias + store (C/D layout: lane n=nn holds rows m=hh*8+r)
    for (int i = 0; i < 2; ++i) {
        for (int t = 0; t < 4; ++t) {
            const int   col  = n0 + t * 16 + nn;
            const float bval = bias[col];
            for (int r = 0; r < 8; ++r) {
                int   m   = m0 + i * 16 + hh * 8 + r;     // global row (b*S+s)
                float val = acc[i][t][r] + bval;
                if (mode == 0) {
                    outF[(size_t)m * D_MODEL + col] = val;
                } else {
                    int b = m >> 11, s = m & (SEQ - 1);
                    int h = col >> 6, d = col & (DK - 1);
                    size_t idx = (mode == 1)
                        ? (((size_t)(b * NHEADS + h) * SEQ + s) * DK + d)
                        : (((size_t)(b * NHEADS + h) * DK + d) * SEQ + s);
                    outH[idx] = f2bf(val);
                }
            }
        }
    }
}

// ---------------------------------------------------------------- flash attention
// Q,K: [B*H][S][DK] bf16 ; Vt: [B*H][DK][S] bf16 ; ctx: [B][S][D_MODEL] bf16
__global__ void flash_attn(const unsigned short* __restrict__ Q,
                           const unsigned short* __restrict__ K,
                           const unsigned short* __restrict__ Vt,
                           const int* __restrict__ mask,
                           unsigned short* __restrict__ ctx) {
    __shared__ __align__(16) unsigned short ldsP[4][16 * 32];  // per-wave P tile [m][k]

    const int lane = threadIdx.x & 31;
    const int w    = threadIdx.x >> 5;
    const int bh   = blockIdx.x;                 // 0..31
    const int b    = bh >> 4, head = bh & 15;
    const int q0   = blockIdx.y * 64 + w * 16;   // 16 query rows per wave
    const int hh   = lane >> 4, nn = lane & 15;

    const unsigned short* Qh = Q  + (size_t)bh * SEQ * DK;
    const unsigned short* Kh = K  + (size_t)bh * SEQ * DK;
    const unsigned short* Vh = Vt + (size_t)bh * DK * SEQ;
    const int* mrow = mask + b * SEQ;

    // resident Q fragments: two 32-deep k-steps covering DK=64
    FragAB qf[2];
    {
        const unsigned short* qRow = Qh + (size_t)(q0 + nn) * DK;
        for (int t = 0; t < 2; ++t) {
            qf[t].q[0] = *(const uint4*)(qRow + t * 32 + hh * 16);
            qf[t].q[1] = *(const uint4*)(qRow + t * 32 + hh * 16 + 8);
        }
    }

    v8f acc[4] = {v8f{}, v8f{}, v8f{}, v8f{}};
    float mrun[8], lrun[8];
    for (int r = 0; r < 8; ++r) { mrun[r] = -1e30f; lrun[r] = 0.0f; }
    const float inv_scale = 0.125f;              // 1/sqrt(DK)

    for (int kb = 0; kb < SEQ; kb += 32) {
        // ---- scores: two 16-key sub-tiles, contraction over DK in 2 WMMA steps
        v8f s0 = {}, s1 = {};
        for (int t = 0; t < 2; ++t) {
            FragAB k0, k1;
            const unsigned short* kR0 = Kh + (size_t)(kb + nn)      * DK + t * 32 + hh * 16;
            const unsigned short* kR1 = Kh + (size_t)(kb + 16 + nn) * DK + t * 32 + hh * 16;
            k0.q[0] = *(const uint4*)(kR0); k0.q[1] = *(const uint4*)(kR0 + 8);
            k1.q[0] = *(const uint4*)(kR1); k1.q[1] = *(const uint4*)(kR1 + 8);
            s0 = __builtin_amdgcn_wmma_f32_16x16x32_bf16(false, qf[t].v, false, k0.v,
                                                         (short)0, s0, false, false);
            s1 = __builtin_amdgcn_wmma_f32_16x16x32_bf16(false, qf[t].v, false, k1.v,
                                                         (short)0, s1, false, false);
        }

        const bool msk0 = mrow[kb + nn] != 0;
        const bool msk1 = mrow[kb + 16 + nn] != 0;
        float p0[8], p1[8], alpha[8];
        // ---- online softmax (row stats replicated across the 16 lanes of each half)
        for (int r = 0; r < 8; ++r) {
            float a = msk0 ? s0[r] * inv_scale : -1e30f;
            float c = msk1 ? s1[r] * inv_scale : -1e30f;
            p0[r] = a; p1[r] = c;
            float t = fmaxf(a, c);
            for (int off = 8; off >= 1; off >>= 1)
                t = fmaxf(t, __shfl_xor(t, off, 16));
            float mnew = fmaxf(mrun[r], t);
            alpha[r] = __expf(mrun[r] - mnew);
            mrun[r]  = mnew;
            p0[r] = __expf(p0[r] - mnew);
            p1[r] = __expf(p1[r] - mnew);
            float rs = p0[r] + p1[r];
            for (int off = 8; off >= 1; off >>= 1)
                rs += __shfl_xor(rs, off, 16);
            lrun[r] = lrun[r] * alpha[r] + rs;
        }
        for (int r = 0; r < 8; ++r) {
            acc[0][r] *= alpha[r]; acc[1][r] *= alpha[r];
            acc[2][r] *= alpha[r]; acc[3][r] *= alpha[r];
        }

        // ---- route P through LDS: C/D layout -> A-matrix layout [m][k(32 keys)]
        unsigned short* P = ldsP[w];
        for (int r = 0; r < 8; ++r) {
            int m = hh * 8 + r;
            P[m * 32 + nn]      = f2bf(p0[r]);
            P[m * 32 + 16 + nn] = f2bf(p1[r]);
        }
        asm volatile("" ::: "memory");   // keep stores before the punned reload
        FragAB pa;
        pa.q[0] = *(const uint4*)(P + nn * 32 + hh * 16);
        pa.q[1] = *(const uint4*)(P + nn * 32 + hh * 16 + 8);
        asm volatile("" ::: "memory");   // keep reload before next iteration's stores

        // ---- PV: 4 d-tiles of 16; Vt gives contiguous per-lane B-frags
        for (int t = 0; t < 4; ++t) {
            FragAB vf;
            const unsigned short* vR = Vh + (size_t)(t * 16 + nn) * SEQ + kb + hh * 16;
            vf.q[0] = *(const uint4*)(vR);
            vf.q[1] = *(const uint4*)(vR + 8);
            acc[t] = __builtin_amdgcn_wmma_f32_16x16x32_bf16(false, pa.v, false, vf.v,
                                                             (short)0, acc[t], false, false);
        }
    }

    // ---- normalize and emit ctx (bf16, [b][s][h*64+d])
    for (int r = 0; r < 8; ++r) {
        float invl = 1.0f / lrun[r];
        int s = q0 + hh * 8 + r;
        size_t base = ((size_t)b * SEQ + s) * D_MODEL + head * DK;
        for (int t = 0; t < 4; ++t)
            ctx[base + t * 16 + nn] = f2bf(acc[t][r] * invl);
    }
}

// ---------------------------------------------------------------- launcher
extern "C" void kernel_launch(void* const* d_in, const int* in_sizes, int n_in,
                              void* d_out, int out_size, void* d_ws, size_t ws_size,
                              hipStream_t stream) {
    (void)in_sizes; (void)n_in; (void)out_size; (void)ws_size;
    const float* x  = (const float*)d_in[0];
    const int* mask = (const int*)d_in[1];
    const float* wq = (const float*)d_in[2];
    const float* bq = (const float*)d_in[3];
    const float* wk = (const float*)d_in[4];
    const float* bk = (const float*)d_in[5];
    const float* wv = (const float*)d_in[6];
    const float* bv = (const float*)d_in[7];
    const float* wo = (const float*)d_in[8];
    const float* bo = (const float*)d_in[9];
    float* out = (float*)d_out;

    // workspace layout (48 MB total)
    char* ws = (char*)d_ws;
    const size_t MB = (size_t)1 << 20;
    unsigned short* xb  = (unsigned short*)(ws + 0 * MB);   // 8 MB  x bf16
    unsigned short* wqT = (unsigned short*)(ws + 8 * MB);   // 2 MB each
    unsigned short* wkT = (unsigned short*)(ws + 10 * MB);
    unsigned short* wvT = (unsigned short*)(ws + 12 * MB);
    unsigned short* woT = (unsigned short*)(ws + 14 * MB);
    unsigned short* Qb  = (unsigned short*)(ws + 16 * MB);  // 8 MB  [b,h,s,d]
    unsigned short* Kb  = (unsigned short*)(ws + 24 * MB);  // 8 MB  [b,h,s,d]
    unsigned short* Vb  = (unsigned short*)(ws + 32 * MB);  // 8 MB  [b,h,d,s]
    unsigned short* Cb  = (unsigned short*)(ws + 40 * MB);  // 8 MB  ctx bf16

    cast_f32_bf16<<<(MROWS * D_MODEL) / 256, 256, 0, stream>>>(x, xb, MROWS * D_MODEL);
    transpose_cast<<<(D_MODEL * D_MODEL) / 256, 256, 0, stream>>>(wq, wqT);
    transpose_cast<<<(D_MODEL * D_MODEL) / 256, 256, 0, stream>>>(wk, wkT);
    transpose_cast<<<(D_MODEL * D_MODEL) / 256, 256, 0, stream>>>(wv, wvT);
    transpose_cast<<<(D_MODEL * D_MODEL) / 256, 256, 0, stream>>>(wo, woT);

    // (4096/32) m-blocks * (1024/64) n-blocks = 2048 waves; 4 waves/block
    const int gemmBlocks = (MROWS / 32) * (D_MODEL / 64) / 4;
    gemm_wmma<<<gemmBlocks, 128, 0, stream>>>(xb, wqT, bq, nullptr, Qb, 1);
    gemm_wmma<<<gemmBlocks, 128, 0, stream>>>(xb, wkT, bk, nullptr, Kb, 1);
    gemm_wmma<<<gemmBlocks, 128, 0, stream>>>(xb, wvT, bv, nullptr, Vb, 2);

    flash_attn<<<dim3(BATCH * NHEADS, SEQ / 64), 128, 0, stream>>>(Qb, Kb, Vb, mask, Cb);

    gemm_wmma<<<gemmBlocks, 128, 0, stream>>>(Cb, woT, bo, out, nullptr, 0);
}
